// AttnBlock_86955907875485
// MI455X (gfx1250) — matrix operand order repaired and checked by reading, not compiled, not run on hardware
//
#include <hip/hip_runtime.h>
#include <hip/hip_bf16.h>

#define B_ 4
#define N_ 4096
#define C_ 512
#define G_ 32

typedef _Float16 v16h __attribute__((ext_vector_type(16)));
typedef float    v8f  __attribute__((ext_vector_type(8)));
typedef _Float16 h2   __attribute__((ext_vector_type(2)));

__device__ __forceinline__ v8f zero8() {
    v8f z;
#pragma unroll
    for (int i = 0; i < 8; ++i) z[i] = 0.0f;
    return z;
}

// A fragment: 16x32 f16, element (m,k). Lanes 0-15: m=lane, k=0..15;
// lanes 16-31: m=lane-16, k=16..31. VGPR i holds k pair (2 f16 / dword).
__device__ __forceinline__ v16h frag_a(const _Float16* __restrict__ src, int ld, int lane) {
    const int hf = lane >> 4, m = lane & 15;
    const _Float16* p = src + (size_t)m * ld + hf * 16;
    v16h a;
#pragma unroll
    for (int i = 0; i < 8; ++i) {
        h2 x = *(const h2*)(p + 2 * i);        // k = 16*hf + 2i
        a[2 * i] = x[0]; a[2 * i + 1] = x[1];
    }
    return a;
}

// B fragment: 32x16 f16, element (k,n) = src[n*ld + k] (src is N-major / transposed).
// Lanes 0-15: n=lane, k=0..15; lanes 16-31: n=lane-16, k=16..31. Contiguous per lane.
__device__ __forceinline__ v16h frag_b(const _Float16* __restrict__ src, int ld, int lane) {
    const int hf = lane >> 4, n = lane & 15;
    const _Float16* p = src + (size_t)n * ld + hf * 16;
    v16h b;
#pragma unroll
    for (int i = 0; i < 8; ++i) {
        h2 x = *(const h2*)(p + 2 * i);
        b[2 * i] = x[0]; b[2 * i + 1] = x[1];
    }
    return b;
}

__device__ __forceinline__ v8f wmma16(v16h a, v16h b, v8f c) {
    return __builtin_amdgcn_wmma_f32_16x16x32_f16(false, a, false, b, (short)0, c, false, false);
}

// ---------------- GroupNorm ----------------
__global__ __launch_bounds__(256) void gn_stats(const float* __restrict__ x,
                                                float* __restrict__ stats) {
    __shared__ float s_sum[256], s_sq[256];
    const int b = blockIdx.x >> 5, g = blockIdx.x & 31;
    const int tid = threadIdx.x;
    float sum = 0.f, sq = 0.f;
    for (int n = tid; n < N_; n += 256) {
        const float4* p = (const float4*)(x + ((size_t)b * N_ + n) * C_ + g * 16);
#pragma unroll
        for (int i = 0; i < 4; ++i) {
            float4 v = p[i];
            sum += v.x + v.y + v.z + v.w;
            sq  += v.x * v.x + v.y * v.y + v.z * v.z + v.w * v.w;
        }
    }
    s_sum[tid] = sum; s_sq[tid] = sq;
    __syncthreads();
    for (int s = 128; s > 0; s >>= 1) {
        if (tid < s) { s_sum[tid] += s_sum[tid + s]; s_sq[tid] += s_sq[tid + s]; }
        __syncthreads();
    }
    if (tid == 0) {
        const float cnt = (float)(N_ * 16);
        float mean = s_sum[0] / cnt;
        float var  = s_sq[0] / cnt - mean * mean;
        stats[2 * blockIdx.x]     = mean;
        stats[2 * blockIdx.x + 1] = rsqrtf(var + 1e-6f);
    }
}

__global__ __launch_bounds__(256) void gn_norm(const float* __restrict__ x,
                                               const float* __restrict__ stats,
                                               const float* __restrict__ gamma,
                                               const float* __restrict__ beta,
                                               _Float16* __restrict__ hn) {
    const size_t idx = (size_t)blockIdx.x * 256 + threadIdx.x;  // 4 channels each
    const size_t e = idx * 4;
    if (e >= (size_t)B_ * N_ * C_) return;
    const int c = (int)(e % C_);
    const int b = (int)(e / ((size_t)N_ * C_));
    const int grp = b * G_ + (c >> 4);
    const float mean = stats[2 * grp], rstd = stats[2 * grp + 1];
    float4 v = *(const float4*)(x + e);
    hn[e + 0] = (_Float16)((v.x - mean) * rstd * gamma[c + 0] + beta[c + 0]);
    hn[e + 1] = (_Float16)((v.y - mean) * rstd * gamma[c + 1] + beta[c + 1]);
    hn[e + 2] = (_Float16)((v.z - mean) * rstd * gamma[c + 2] + beta[c + 2]);
    hn[e + 3] = (_Float16)((v.w - mean) * rstd * gamma[c + 3] + beta[c + 3]);
}

// Wt[n][k] = (f16) W[k][n] : transposed so WMMA B-fragments load contiguously.
__global__ __launch_bounds__(256) void wconvert(const float* __restrict__ W,
                                                _Float16* __restrict__ Wt) {
    const int idx = blockIdx.x * 256 + threadIdx.x;
    if (idx >= C_ * C_) return;
    const int n = idx / C_, kx = idx % C_;
    Wt[idx] = (_Float16)W[(size_t)kx * C_ + n];
}

// ---------------- WMMA GEMM: [M][512] f16 @ Wt[512][512] f16 + bias -> f16 ----------------
// 64x64 output tile per WG, 8 waves * 2 tiles (shared A frag per wave pair).
// transposedStore: write Out[b][col][n] (used to pre-transpose V for attention).
__global__ __launch_bounds__(256) void gemm_f16(const _Float16* __restrict__ A,
                                                const _Float16* __restrict__ Bt,
                                                const float* __restrict__ bias,
                                                _Float16* __restrict__ Out,
                                                int transposedStore) {
    const int tid = threadIdx.x, wave = tid >> 5, lane = tid & 31;
    const int hf = lane >> 4, nl = lane & 15;
    const int row0 = blockIdx.y * 64, col0 = blockIdx.x * 64;
    const int mt = (wave * 2) >> 2, nt0 = (wave * 2) & 3;
    v8f acc0 = zero8(), acc1 = zero8();
    const _Float16* arow  = A  + (size_t)(row0 + mt * 16) * C_;
    const _Float16* brow0 = Bt + (size_t)(col0 + nt0 * 16) * C_;
    const _Float16* brow1 = Bt + (size_t)(col0 + nt0 * 16 + 16) * C_;
#pragma unroll 4
    for (int kk = 0; kk < C_; kk += 32) {
        v16h a = frag_a(arow + kk, C_, lane);
        acc0 = wmma16(a, frag_b(brow0 + kk, C_, lane), acc0);
        acc1 = wmma16(a, frag_b(brow1 + kk, C_, lane), acc1);
    }
#pragma unroll
    for (int i = 0; i < 2; ++i) {
        v8f acc = i ? acc1 : acc0;
        const int col = col0 + (nt0 + i) * 16 + nl;
        const float bv = bias[col];
#pragma unroll
        for (int r = 0; r < 8; ++r) {
            const int row = row0 + mt * 16 + hf * 8 + r;
            const float val = acc[r] + bv;
            if (transposedStore)
                Out[((size_t)(row >> 12) * C_ + col) * N_ + (row & (N_ - 1))] = (_Float16)val;
            else
                Out[(size_t)row * C_ + col] = (_Float16)val;
        }
    }
}

// ---------------- Flash attention: per (batch, 64-query block) ----------------
// q,k: [b][n][c] f16 ; vT: [b][c][n] f16 ; o: [b][n][c] f16
__global__ __launch_bounds__(256) void attn_fa(const _Float16* __restrict__ q,
                                               const _Float16* __restrict__ k,
                                               const _Float16* __restrict__ vT,
                                               _Float16* __restrict__ o) {
    __shared__ float    Sf[64 * 64];   // scaled scores (f32)
    __shared__ _Float16 Pf[64 * 64];   // exp probabilities (f16) = next A operand
    __shared__ float row_max[64], row_sum[64], row_alpha[64];

    const int tid = threadIdx.x, wave = tid >> 5, lane = tid & 31;
    const int hf = lane >> 4, nl = lane & 15;
    const int qb = blockIdx.x, b = blockIdx.y;
    const size_t base = (size_t)b * N_ * C_;
    const float scale = 0.044194173824159216f;  // 512^-0.5

    if (tid < 64) { row_max[tid] = -3.0e38f; row_sum[tid] = 0.0f; }
    v8f acc[4][4];  // wave owns O[:, wave*64 .. wave*64+64) : 4x4 tiles of 16x16
#pragma unroll
    for (int mtI = 0; mtI < 4; ++mtI)
#pragma unroll
        for (int ntI = 0; ntI < 4; ++ntI) acc[mtI][ntI] = zero8();
    __syncthreads();

    for (int kb = 0; kb < N_ / 64; ++kb) {
        // ---- S = scale * Q Kt : 16 tiles over 8 waves (2 each, shared A) ----
        {
            const int mt = (wave * 2) >> 2, nt0 = (wave * 2) & 3;
            v8f s0 = zero8(), s1 = zero8();
            const _Float16* arow  = q + base + (size_t)(qb * 64 + mt * 16) * C_;
            const _Float16* brow0 = k + base + (size_t)(kb * 64 + nt0 * 16) * C_;
            const _Float16* brow1 = k + base + (size_t)(kb * 64 + nt0 * 16 + 16) * C_;
#pragma unroll 4
            for (int kk = 0; kk < C_; kk += 32) {
                v16h a = frag_a(arow + kk, C_, lane);
                s0 = wmma16(a, frag_b(brow0 + kk, C_, lane), s0);
                s1 = wmma16(a, frag_b(brow1 + kk, C_, lane), s1);
            }
#pragma unroll
            for (int r = 0; r < 8; ++r) {
                const int rr = (mt * 16 + hf * 8 + r) * 64;
                Sf[rr + nt0 * 16 + nl]      = s0[r] * scale;
                Sf[rr + nt0 * 16 + 16 + nl] = s1[r] * scale;
            }
        }
        __syncthreads();

        // ---- online softmax, one row per thread (threads 0..63) ----
        if (tid < 64) {
            const float m_prev = row_max[tid], l_prev = row_sum[tid];
            float mx = m_prev;
            const float* srow = &Sf[tid * 64];
            for (int j = 0; j < 64; ++j) mx = fmaxf(mx, srow[j]);
            const float alpha = __expf(m_prev - mx);
            float s = 0.0f;
            _Float16* prow = &Pf[tid * 64];
            for (int j = 0; j < 64; ++j) {
                const float p = __expf(srow[j] - mx);
                s += p;
                prow[j] = (_Float16)p;
            }
            row_max[tid]   = mx;
            row_sum[tid]   = l_prev * alpha + s;
            row_alpha[tid] = alpha;
        }
        __syncthreads();

        // ---- rescale running O, then O += P @ V ----
#pragma unroll
        for (int mtI = 0; mtI < 4; ++mtI) {
            float al[8];
#pragma unroll
            for (int r = 0; r < 8; ++r) al[r] = row_alpha[mtI * 16 + hf * 8 + r];
#pragma unroll
            for (int ntI = 0; ntI < 4; ++ntI)
#pragma unroll
                for (int r = 0; r < 8; ++r) acc[mtI][ntI][r] *= al[r];
        }
#pragma unroll
        for (int kk = 0; kk < 2; ++kk) {
            v16h afr[4];
#pragma unroll
            for (int mtI = 0; mtI < 4; ++mtI)
                afr[mtI] = frag_a(&Pf[(mtI * 16) * 64 + kk * 32], 64, lane);
#pragma unroll
            for (int ntI = 0; ntI < 4; ++ntI) {
                v16h bfr = frag_b(vT + ((size_t)b * C_ + wave * 64 + ntI * 16) * N_ +
                                      kb * 64 + kk * 32,
                                  N_, lane);
#pragma unroll
                for (int mtI = 0; mtI < 4; ++mtI)
                    acc[mtI][ntI] = wmma16(afr[mtI], bfr, acc[mtI][ntI]);
            }
        }
        __syncthreads();
    }

    // ---- normalize by row sums, store O (f16) ----
#pragma unroll
    for (int mtI = 0; mtI < 4; ++mtI) {
        float inv[8];
#pragma unroll
        for (int r = 0; r < 8; ++r) inv[r] = 1.0f / row_sum[mtI * 16 + hf * 8 + r];
#pragma unroll
        for (int ntI = 0; ntI < 4; ++ntI) {
            const int col = wave * 64 + ntI * 16 + nl;
#pragma unroll
            for (int r = 0; r < 8; ++r) {
                const int row = qb * 64 + mtI * 16 + hf * 8 + r;
                o[base + (size_t)row * C_ + col] = (_Float16)(acc[mtI][ntI][r] * inv[r]);
            }
        }
    }
}

// ---------------- out projection + bias + residual -> f32 ----------------
__global__ __launch_bounds__(256) void gemm_proj_out(const _Float16* __restrict__ A,
                                                     const _Float16* __restrict__ Bt,
                                                     const float* __restrict__ bias,
                                                     const float* __restrict__ resid,
                                                     float* __restrict__ Out) {
    const int tid = threadIdx.x, wave = tid >> 5, lane = tid & 31;
    const int hf = lane >> 4, nl = lane & 15;
    const int row0 = blockIdx.y * 64, col0 = blockIdx.x * 64;
    const int mt = (wave * 2) >> 2, nt0 = (wave * 2) & 3;
    v8f acc0 = zero8(), acc1 = zero8();
    const _Float16* arow  = A  + (size_t)(row0 + mt * 16) * C_;
    const _Float16* brow0 = Bt + (size_t)(col0 + nt0 * 16) * C_;
    const _Float16* brow1 = Bt + (size_t)(col0 + nt0 * 16 + 16) * C_;
#pragma unroll 4
    for (int kk = 0; kk < C_; kk += 32) {
        v16h a = frag_a(arow + kk, C_, lane);
        acc0 = wmma16(a, frag_b(brow0 + kk, C_, lane), acc0);
        acc1 = wmma16(a, frag_b(brow1 + kk, C_, lane), acc1);
    }
#pragma unroll
    for (int i = 0; i < 2; ++i) {
        v8f acc = i ? acc1 : acc0;
        const int col = col0 + (nt0 + i) * 16 + nl;
        const float bv = bias[col];
#pragma unroll
        for (int r = 0; r < 8; ++r) {
            const size_t row = (size_t)(row0 + mt * 16 + hf * 8 + r);
            Out[row * C_ + col] = acc[r] + bv + resid[row * C_ + col];
        }
    }
}

extern "C" void kernel_launch(void* const* d_in, const int* in_sizes, int n_in,
                              void* d_out, int out_size, void* d_ws, size_t ws_size,
                              hipStream_t stream) {
    (void)in_sizes; (void)n_in; (void)out_size; (void)ws_size;
    const float* x        = (const float*)d_in[0];
    const float* gn_scale = (const float*)d_in[1];
    const float* gn_bias  = (const float*)d_in[2];
    const float* wq = (const float*)d_in[3];
    const float* bq = (const float*)d_in[4];
    const float* wk = (const float*)d_in[5];
    const float* bk = (const float*)d_in[6];
    const float* wv = (const float*)d_in[7];
    const float* bv = (const float*)d_in[8];
    const float* wo = (const float*)d_in[9];
    const float* bo = (const float*)d_in[10];
    float* out = (float*)d_out;

    char* ws = (char*)d_ws;
    size_t off = 0;
    auto alloc = [&](size_t bytes) {
        char* p = ws + off;
        off += (bytes + 255) & ~(size_t)255;
        return p;
    };
    const size_t tok = (size_t)B_ * N_ * C_;
    float*    stats = (float*)alloc((size_t)2 * B_ * G_ * sizeof(float));
    _Float16* hn  = (_Float16*)alloc(tok * 2);
    _Float16* d_q = (_Float16*)alloc(tok * 2);
    _Float16* d_k = (_Float16*)alloc(tok * 2);
    _Float16* d_vT = (_Float16*)alloc(tok * 2);   // [b][c][n]
    _Float16* d_o = (_Float16*)alloc(tok * 2);
    _Float16* wqt = (_Float16*)alloc((size_t)C_ * C_ * 2);
    _Float16* wkt = (_Float16*)alloc((size_t)C_ * C_ * 2);
    _Float16* wvt = (_Float16*)alloc((size_t)C_ * C_ * 2);
    _Float16* wot = (_Float16*)alloc((size_t)C_ * C_ * 2);

    gn_stats<<<B_ * G_, 256, 0, stream>>>(x, stats);
    gn_norm<<<(unsigned)(tok / 4 / 256), 256, 0, stream>>>(x, stats, gn_scale, gn_bias, hn);

    const int wblk = (C_ * C_ + 255) / 256;
    wconvert<<<wblk, 256, 0, stream>>>(wq, wqt);
    wconvert<<<wblk, 256, 0, stream>>>(wk, wkt);
    wconvert<<<wblk, 256, 0, stream>>>(wv, wvt);
    wconvert<<<wblk, 256, 0, stream>>>(wo, wot);

    dim3 gGemm(C_ / 64, (B_ * N_) / 64);
    gemm_f16<<<gGemm, 256, 0, stream>>>(hn, wqt, bq, d_q, 0);
    gemm_f16<<<gGemm, 256, 0, stream>>>(hn, wkt, bk, d_k, 0);
    gemm_f16<<<gGemm, 256, 0, stream>>>(hn, wvt, bv, d_vT, 1);

    dim3 gAttn(N_ / 64, B_);
    attn_fa<<<gAttn, 256, 0, stream>>>(d_q, d_k, d_vT, d_o);

    gemm_proj_out<<<gGemm, 256, 0, stream>>>(d_o, wot, bo, x, out);
}
